// CLoRALinear_28853590294649
// MI455X (gfx1250) — matrix operand order repaired and checked by reading, not compiled, not run on hardware
//
#include <hip/hip_runtime.h>
#include <hip/hip_bf16.h>

// ---------------------------------------------------------------------------
// LoRA linear: y = x @ (W + A@B^T * scale)^T + bias
//   x: [4,2048,4096] f32   W: [4096,4096] f32   bias: [4096] f32
//   A: [4096,32] f32       B: [4096,32] f32     scale = 32/32 = 1.0
// Strategy: fold LoRA into bf16 W_eff once, convert x to bf16 once, then a
// single bf16-WMMA GEMM (f32 accumulate) with bias epilogue.
// This round: global->LDS staging uses GLOBAL_LOAD_ASYNC_TO_LDS_B128
// (ASYNCcnt-tracked, no VGPR round-trip), triple-buffered LDS pipeline.
// ---------------------------------------------------------------------------

#define IN_DIM   4096
#define OUT_DIM  4096
#define M_ROWS   8192          // 4 * 2048
#define RANK     32
#define LORA_SCALE 1.0f        // alpha / r = 32/32

typedef __attribute__((ext_vector_type(16))) __bf16 v16bf;
typedef __attribute__((ext_vector_type(4)))  __bf16 v4bf;
typedef __attribute__((ext_vector_type(8)))  float  v8f;

union FragU {                  // 16 bf16 = 32 bytes = 8 VGPRs
    v16bf v;
    uint4 q[2];
};

// ---------------------------------------------------------------------------
// Kernel 1: W_eff[o][k] = bf16( W[o][k] + scale * dot(A[o,:], B[k,:]) )
// ---------------------------------------------------------------------------
__global__ __launch_bounds__(256) void lora_fold_weff_kernel(
    const float* __restrict__ W, const float* __restrict__ A,
    const float* __restrict__ Bm, __bf16* __restrict__ weff)
{
    const size_t idx = (size_t)blockIdx.x * 256 + threadIdx.x; // o*4096 + k
    const int o = (int)(idx >> 12);
    const int k = (int)(idx & 4095);
    const float* ar = A  + (size_t)o * RANK;
    const float* br = Bm + (size_t)k * RANK;
    float s = 0.f;
#pragma unroll
    for (int r = 0; r < RANK; ++r) s += ar[r] * br[r];
    weff[idx] = (__bf16)(W[idx] + s * LORA_SCALE);
}

// ---------------------------------------------------------------------------
// Kernel 2: x (f32) -> x (bf16), 4 elements per thread
// ---------------------------------------------------------------------------
__global__ __launch_bounds__(256) void lora_xconv_kernel(
    const float* __restrict__ x, __bf16* __restrict__ xb)
{
    const size_t i = (size_t)blockIdx.x * 256 + threadIdx.x;   // float4 index
    const float4 v = ((const float4*)x)[i];
    v4bf h = { (__bf16)v.x, (__bf16)v.y, (__bf16)v.z, (__bf16)v.w };
    ((v4bf*)xb)[i] = h;
}

// ---------------------------------------------------------------------------
// Kernel 3: C[m][o] = sum_k Xb[m][k] * Weff[o][k] + bias[o]
// Block tile 128x128, K-step 32, 256 threads = 8 waves (2 M x 4 N),
// each wave: 64x32 patch = 4x2 accumulators of v_wmma_f32_16x16x32_bf16.
// Staging: async global->LDS b128, triple buffer, depth-2 pipeline.
// ---------------------------------------------------------------------------
#define TM 128
#define TN 128
#define TK 32
#define NBUF 3
#define LDS_STRIDE 40          // 32 + 8 pad (80B rows, 16B aligned)

__global__ __launch_bounds__(256) void lora_gemm_bf16_kernel(
    const __bf16* __restrict__ xb,      // [M_ROWS][IN_DIM]
    const __bf16* __restrict__ weff,    // [OUT_DIM][IN_DIM]
    const float*  __restrict__ bias,    // [OUT_DIM]
    float* __restrict__ out)            // [M_ROWS][OUT_DIM]
{
    __shared__ __bf16 lx[NBUF][TM][LDS_STRIDE];  // x tile    [m][k]
    __shared__ __bf16 lw[NBUF][TN][LDS_STRIDE];  // Weff tile [n][k]

    const int t       = threadIdx.x;
    const int lane    = t & 31;
    const int wid     = t >> 5;        // 0..7
    const int wave_m  = wid >> 2;      // 0..1  (64 rows each)
    const int wave_n  = wid & 3;       // 0..3  (32 cols each)
    const int l16     = lane & 15;
    const int half    = lane >> 4;     // 0 or 1

    const int n_block = blockIdx.x * TN;
    const int m_block = blockIdx.y * TM;

    // staging: thread -> (row t>>1, 16-element chunk (t&1)*16); 32B per tile
    const int srow  = t >> 1;          // 0..127
    const int skoff = (t & 1) * 16;    // element offset within K-row

    // Per-thread constant byte offsets of the staged row starts
    const unsigned gx_row = (unsigned)(m_block + srow) * (IN_DIM * 2u);
    const unsigned gw_row = (unsigned)(n_block + srow) * (IN_DIM * 2u);

    // 4 x GLOBAL_LOAD_ASYNC_TO_LDS_B128 per thread per K-step.
    // IOFFSET is added to BOTH the LDS and global addresses (ISA 08 4.4),
    // so offset:0 / offset:16 covers the 32B chunk from one address pair.
    auto stage = [&](int ks, int buf) {
        const unsigned kb = (unsigned)(ks * TK + skoff) * 2u;   // byte col
        const unsigned vx = gx_row + kb;
        const unsigned vw = gw_row + kb;
        const unsigned lxa = (unsigned)(uintptr_t)&lx[buf][srow][skoff];
        const unsigned lwa = (unsigned)(uintptr_t)&lw[buf][srow][skoff];
        asm volatile(
            "global_load_async_to_lds_b128 %0, %2, %4 offset:0\n\t"
            "global_load_async_to_lds_b128 %0, %2, %4 offset:16\n\t"
            "global_load_async_to_lds_b128 %1, %3, %5 offset:0\n\t"
            "global_load_async_to_lds_b128 %1, %3, %5 offset:16"
            :: "v"(lxa), "v"(lwa), "v"(vx), "v"(vw), "s"(xb), "s"(weff)
            : "memory");
    };

    v8f acc[4][2];
    const v8f vzero = {0.f, 0.f, 0.f, 0.f, 0.f, 0.f, 0.f, 0.f};
#pragma unroll
    for (int mt = 0; mt < 4; ++mt)
#pragma unroll
        for (int nt = 0; nt < 2; ++nt) acc[mt][nt] = vzero;

    const int NK = IN_DIM / TK;        // 128 K-steps
    stage(0, 0);
    stage(1, 1);

    for (int ks = 0; ks < NK; ++ks) {
        const int cur = ks % NBUF;

        // Own wave's stage(ks) is the oldest in flight: <=4 leaves only
        // stage(ks+1) outstanding. Barrier then makes it block-visible.
        if (ks + 1 < NK) { asm volatile("s_wait_asynccnt 0x4" ::: "memory"); }
        else             { asm volatile("s_wait_asynccnt 0x0" ::: "memory"); }
        __syncthreads();

        // Safe to refill buf (ks+2)%3: its last readers ran in iteration
        // ks-1, and every wave passed the barrier above since then.
        if (ks + 2 < NK) stage(ks + 2, (ks + 2) % NBUF);

        // Load fragments (layouts match ISA 16-bit A/B VGPR striping).
        FragU af[4];
#pragma unroll
        for (int mt = 0; mt < 4; ++mt) {
            const __bf16* p = &lx[cur][wave_m * 64 + mt * 16 + l16][half * 8];
            af[mt].q[0] = *(const uint4*)(p);        // K = h*8 .. h*8+7
            af[mt].q[1] = *(const uint4*)(p + 16);   // K = h*8+16 .. +23
        }
        FragU bf[2];
#pragma unroll
        for (int nt = 0; nt < 2; ++nt) {
            const __bf16* p = &lw[cur][wave_n * 32 + nt * 16 + l16][half * 8];
            bf[nt].q[0] = *(const uint4*)(p);        // K = h*8 .. h*8+7
            bf[nt].q[1] = *(const uint4*)(p + 16);   // K = h*8+16 .. +23
        }

#pragma unroll
        for (int mt = 0; mt < 4; ++mt)
#pragma unroll
            for (int nt = 0; nt < 2; ++nt)
                acc[mt][nt] = __builtin_amdgcn_wmma_f32_16x16x32_bf16(
                    false, af[mt].v, false, bf[nt].v,
                    (short)0, acc[mt][nt], false, false);
    }

    // Epilogue: C VGPR j -> row j (lanes 0-15) / row j+8 (lanes 16-31)
#pragma unroll
    for (int mt = 0; mt < 4; ++mt) {
#pragma unroll
        for (int nt = 0; nt < 2; ++nt) {
            const int col   = n_block + wave_n * 32 + nt * 16 + l16;
            const float bv  = bias[col];
            const int rbase = m_block + wave_m * 64 + mt * 16 + half * 8;
#pragma unroll
            for (int j = 0; j < 8; ++j)
                out[(size_t)(rbase + j) * OUT_DIM + col] = acc[mt][nt][j] + bv;
        }
    }
}

// ---------------------------------------------------------------------------
extern "C" void kernel_launch(void* const* d_in, const int* in_sizes, int n_in,
                              void* d_out, int out_size, void* d_ws, size_t ws_size,
                              hipStream_t stream) {
    const float* x    = (const float*)d_in[0];
    const float* W    = (const float*)d_in[1];
    const float* bias = (const float*)d_in[2];
    const float* A    = (const float*)d_in[3];
    const float* Bm   = (const float*)d_in[4];
    float* out        = (float*)d_out;

    // Workspace layout: [0, 33.5MB) W_eff bf16 ; [33.5MB, 100.7MB) x bf16
    __bf16* weff = (__bf16*)d_ws;
    __bf16* xb   = (__bf16*)((char*)d_ws + (size_t)OUT_DIM * IN_DIM * sizeof(__bf16));

    // 1) Fold LoRA into bf16 W_eff
    lora_fold_weff_kernel<<<(OUT_DIM * IN_DIM) / 256, 256, 0, stream>>>(W, A, Bm, weff);
    // 2) Convert x to bf16 (4 elems/thread)
    lora_xconv_kernel<<<((size_t)M_ROWS * IN_DIM / 4) / 256, 256, 0, stream>>>(x, xb);
    // 3) Main GEMM + bias
    dim3 grid(OUT_DIM / TN, M_ROWS / TM);   // (32, 64)
    lora_gemm_bf16_kernel<<<grid, 256, 0, stream>>>(xb, weff, bias, out);
}
